// I2TCrossAttentionPool_70231305224548
// MI455X (gfx1250) — matrix operand order repaired and checked by reading, not compiled, not run on hardware
//
#include <hip/hip_runtime.h>
#include <hip/hip_bf16.h>

// ---------------------------------------------------------------------------
// SCAN i2t cross-attention pooling for MI455X (gfx1250, wave32, WMMA).
//   out[i,c] = mean_{r<len_i} cos( img[i,r], wctx[c,r] )  with SCAN softmax.
// Key identities (avoid materializing wctx):
//   s1[c,r] = sum_w p[c,r,w] * raw[c,w,r]          (raw = caps . img, pre-leaky)
//   s2[c,r] = p^T G_c p,  G_c = caps_c caps_c^T    (per-caption Gram, WMMA)
// CDNA5 paths: v_wmma_f32_16x16x32_bf16 for both contractions,
//              global_load_async_to_lds_b128 (+ s_wait_asynccnt) for staging.
// ---------------------------------------------------------------------------

typedef __attribute__((ext_vector_type(16))) __bf16 v16bf;
typedef __attribute__((ext_vector_type(8)))  float  v8f;

#define N_IMG   512
#define R_REG   36
#define RP      48          // regions padded to 3 WMMA N-tiles
#define W_WORD  50
#define D_DIM   1024
#define NC      8           // captions per workgroup (== 8 waves)
#define MROWS   (NC * W_WORD)   // 400 rows == 25 exact M-tiles
#define MTILES  25
#define GDIM    64          // Gram padded to 64x64
#define SMOOTH  9.0f
#define EPS_F   1e-8f

// LDS strides chosen for 16B alignment + conflict-free b128 access
#define IST   1032          // img lds row stride (bf16 elems): 2064B = 129*16
#define RST   49            // raw row stride (f32), odd -> conflict free
#define PBST  72            // p row stride (bf16 elems): 144B = 9*16

#define LDS_IMG   (RP * IST * 2)        // 99072 B (aliased with pbf: 55296 B)
#define LDS_RAW   (MROWS * RST * 4)     // 78400 B
#define LDS_TOTAL (LDS_IMG + LDS_RAW + MROWS*4 + NC*RP*4 + NC*RP*4)  // 182144 B

union FragBF { v16bf v; uint4 q[2]; };

static __device__ __forceinline__ v8f wmma_bf16(const FragBF& a, const FragBF& b, v8f c) {
    return __builtin_amdgcn_wmma_f32_16x16x32_bf16(false, a.v, false, b.v,
                                                   (short)0, c, false, false);
}

// Async Global -> LDS copy, 16B per lane; tracked by ASYNCcnt (not DScnt!).
static __device__ __forceinline__ void async_load_b128(unsigned lds_off, const void* gaddr) {
    asm volatile("global_load_async_to_lds_b128 %0, %1, off"
                 :: "v"(lds_off), "v"(gaddr) : "memory");
}
static __device__ __forceinline__ void wait_asynccnt0() {
    asm volatile("s_wait_asynccnt 0x0" ::: "memory");
}

// ---------------------------------------------------------------- convert ---
extern "C" __global__ void scan_convert(const float* __restrict__ caps,
                                        const float* __restrict__ imgs,
                                        __hip_bfloat16* __restrict__ caps_bf,
                                        __hip_bfloat16* __restrict__ imgs_bf,
                                        int capsN, int imgsN)
{
    const int stride = gridDim.x * blockDim.x;
    for (int i = blockIdx.x * blockDim.x + threadIdx.x; i < capsN; i += stride)
        caps_bf[i] = __float2bfloat16(caps[i]);
    for (int i = blockIdx.x * blockDim.x + threadIdx.x; i < imgsN; i += stride)
        imgs_bf[i] = __float2bfloat16(imgs[i]);
}

// ------------------------------------------------------ region norms (f32) ---
extern "C" __global__ __launch_bounds__(256)
void scan_w1(const float* __restrict__ imgs, float* __restrict__ w1g, int nRows)
{
    const int row  = blockIdx.x * 8 + (threadIdx.x >> 5);
    const int lane = threadIdx.x & 31;
    if (row >= nRows) return;
    const float* p = imgs + (size_t)row * D_DIM;
    float s = 0.f;
    for (int d = lane; d < D_DIM; d += 32) { float x = p[d]; s += x * x; }
#pragma unroll
    for (int m = 16; m >= 1; m >>= 1) s += __shfl_xor(s, m, 32);
    if (lane == 0) w1g[row] = sqrtf(s);
}

// --------------------------------------------- per-caption Gram via WMMA ----
extern "C" __global__ __launch_bounds__(256)
void scan_gram(const __hip_bfloat16* __restrict__ caps_bf,
               __hip_bfloat16* __restrict__ G_bf)
{
    const int c    = blockIdx.x;
    const int wave = threadIdx.x >> 5;
    const int lane = threadIdx.x & 31;
    const int half = lane >> 4;
    const int lrow = lane & 15;
    const size_t base = (size_t)c * W_WORD * D_DIM;

    for (int pair = wave * 2; pair < wave * 2 + 2; ++pair) {   // 16 (mt,nt) pairs
        const int mt = pair >> 2, nt = pair & 3;
        int aRow = mt * 16 + lrow; if (aRow > W_WORD - 1) aRow = W_WORD - 1;
        int bCol = nt * 16 + lrow; if (bCol > W_WORD - 1) bCol = W_WORD - 1;
        v8f acc = {};
        for (int kk = 0; kk < D_DIM; kk += 32) {
            FragBF A, Bf;
            const __hip_bfloat16* ap = caps_bf + base + (size_t)aRow * D_DIM + kk + half * 16;
            A.q[0]  = *((const uint4*)ap);
            A.q[1]  = *((const uint4*)(ap + 8));
            const __hip_bfloat16* bp = caps_bf + base + (size_t)bCol * D_DIM + kk + half * 16;
            Bf.q[0] = *((const uint4*)bp);
            Bf.q[1] = *((const uint4*)(bp + 8));
            acc = wmma_bf16(A, Bf, acc);
        }
#pragma unroll
        for (int v = 0; v < 8; ++v) {
            const int w  = mt * 16 + v + 8 * half;
            const int wp = nt * 16 + lrow;
            const float val = (w < W_WORD && wp < W_WORD) ? acc[v] : 0.f;
            G_bf[((size_t)c * GDIM + w) * GDIM + wp] = __float2bfloat16(val);
        }
    }
}

// --------------------------------------------------------------- main -------
extern "C" __global__ __launch_bounds__(256)
void scan_i2t_main(const __hip_bfloat16* __restrict__ caps_bf,
                   const __hip_bfloat16* __restrict__ imgs_bf,
                   const __hip_bfloat16* __restrict__ G_bf,
                   const float* __restrict__ w1g,
                   const int* __restrict__ img_lens,
                   float* __restrict__ out)
{
    extern __shared__ char smem[];
    __hip_bfloat16* imgLds = (__hip_bfloat16*)smem;                 // [RP][IST]  (phase 1)
    __hip_bfloat16* pbf    = (__hip_bfloat16*)smem;                 // [NC][RP][PBST] (aliased)
    float* raw   = (float*)(smem + LDS_IMG);                        // [MROWS][RST]
    float* norms = (float*)(smem + LDS_IMG + LDS_RAW);              // 9/(||.||+eps) per (c,w)
    float* s1buf = norms + MROWS;                                   // [NC][RP]
    float* s2buf = s1buf + NC * RP;                                 // [NC][RP]

    const int i       = blockIdx.y;
    const int capbase = blockIdx.x * NC;
    const int t    = threadIdx.x;
    const int wave = t >> 5;
    const int lane = t & 31;
    const int half = lane >> 4;
    const int lrow = lane & 15;
    const int len  = img_lens[i];

    // -- phase 0: async-stage image regions (bf16) into LDS, zero pad rows ---
    {
        // dynamic LDS begins right after static LDS: byte base of smem[]
        const unsigned ldsBase = __builtin_amdgcn_groupstaticsize();
        const __hip_bfloat16* gsrc = imgs_bf + (size_t)i * R_REG * D_DIM;
        for (int idx = t; idx < R_REG * 128; idx += 256) {          // 36 rows x 128 uint4
            const int r = idx >> 7, q = idx & 127;
            const unsigned loff = ldsBase + (unsigned)(r * (IST * 2) + q * 16);
            async_load_b128(loff, (const void*)(gsrc + (size_t)r * D_DIM + q * 8));
        }
        const uint4 z = {0, 0, 0, 0};
        for (int idx = R_REG * 128 + t; idx < RP * 128; idx += 256) // pad rows 36..47
        {
            const int r = idx >> 7, q = idx & 127;
            ((uint4*)(imgLds + r * IST))[q] = z;
        }
        wait_asynccnt0();   // ASYNCcnt is NOT fenced by __syncthreads' waitcnts
    }
    __syncthreads();

    // -- phase 1: raw[(c,w), r] = caps . img^T  (bf16 WMMA, K = 1024) --------
    for (int mt = wave; mt < MTILES; mt += 8) {
        const int m16 = mt * 16;
        const size_t aBase = ((size_t)(capbase * W_WORD + m16 + lrow)) * D_DIM;
        v8f acc0 = {}, acc1 = {}, acc2 = {};
        for (int kk = 0; kk < D_DIM; kk += 32) {
            FragBF A, B0, B1, B2;
            const int ad = kk + half * 16;                          // A: contiguous 32B
            A.q[0] = *((const uint4*)(caps_bf + aBase + ad));
            A.q[1] = *((const uint4*)(caps_bf + aBase + ad + 8));
            const int bd = kk + half * 16;                          // B: contiguous 32B
            const __hip_bfloat16* p0 = imgLds + (size_t)(lrow)      * IST + bd;
            const __hip_bfloat16* p1 = imgLds + (size_t)(16 + lrow) * IST + bd;
            const __hip_bfloat16* p2 = imgLds + (size_t)(32 + lrow) * IST + bd;
            B0.q[0] = *((const uint4*)p0); B0.q[1] = *((const uint4*)(p0 + 8));
            B1.q[0] = *((const uint4*)p1); B1.q[1] = *((const uint4*)(p1 + 8));
            B2.q[0] = *((const uint4*)p2); B2.q[1] = *((const uint4*)(p2 + 8));
            acc0 = wmma_bf16(A, B0, acc0);
            acc1 = wmma_bf16(A, B1, acc1);
            acc2 = wmma_bf16(A, B2, acc2);
        }
#pragma unroll
        for (int v = 0; v < 8; ++v) {
            const int m = m16 + v + 8 * half;
            raw[m * RST +      lrow] = acc0[v];
            raw[m * RST + 16 + lrow] = acc1[v];
            raw[m * RST + 32 + lrow] = acc2[v];
        }
    }
    __syncthreads();

    // -- phase 2a: per (c,w) row, 9 / (||lrelu(raw)*mask||_r + eps) ----------
    for (int m = t; m < MROWS; m += 256) {
        float s = 0.f;
        for (int r = 0; r < len; ++r) {
            const float x  = raw[m * RST + r];
            const float xl = x >= 0.f ? x : 0.1f * x;
            s += xl * xl;
        }
        norms[m] = SMOOTH / (sqrtf(s) + EPS_F);
    }
    __syncthreads();

    // -- phase 2b: softmax over words, s1 = sum_w p*raw, write p (bf16) ------
    for (int idx = t; idx < NC * RP; idx += 256) {
        const int c = idx / RP, r = idx % RP;
        __hip_bfloat16* pr = pbf + (size_t)(c * RP + r) * PBST;
        float s1 = 0.f;
        if (r < len) {
            const float* rawc = raw + (c * W_WORD) * RST + r;
            const float* nv   = norms + c * W_WORD;
            float amax = -1e30f;
            for (int w = 0; w < W_WORD; ++w) {
                const float x  = rawc[w * RST];
                const float xl = x >= 0.f ? x : 0.1f * x;
                const float a  = xl * nv[w];
                amax = fmaxf(amax, a);
            }
            float sum = 0.f, s1n = 0.f;
            for (int w = 0; w < W_WORD; ++w) {
                const float x  = rawc[w * RST];
                const float xl = x >= 0.f ? x : 0.1f * x;
                const float e  = __expf(xl * nv[w] - amax);
                sum += e;
                s1n += e * x;                     // uses pre-leaky raw score
            }
            const float inv = 1.f / sum;
            s1 = s1n * inv;
            for (int w = 0; w < W_WORD; ++w) {
                const float x  = rawc[w * RST];
                const float xl = x >= 0.f ? x : 0.1f * x;
                pr[w] = __float2bfloat16(__expf(xl * nv[w] - amax) * inv);
            }
            for (int w = W_WORD; w < GDIM; ++w) pr[w] = __float2bfloat16(0.f);
        } else {
            for (int w = 0; w < GDIM; ++w) pr[w] = __float2bfloat16(0.f);
        }
        s1buf[idx] = s1;
    }
    __syncthreads();

    // -- phase 3: s2[c,r] = p^T G_c p  via  h = p @ G (WMMA) + lane reduce ---
    {
        const int c = wave;                                         // 1 cap per wave
        FragBF g[2][4];
        const size_t gBase = (size_t)(capbase + c) * GDIM * GDIM;
#pragma unroll
        for (int kt = 0; kt < 2; ++kt)
#pragma unroll
            for (int nt = 0; nt < 4; ++nt) {                        // symmetric G: col = row
                const __hip_bfloat16* gp =
                    G_bf + gBase + (size_t)(nt * 16 + lrow) * GDIM + kt * 32 + half * 16;
                g[kt][nt].q[0] = *((const uint4*)gp);
                g[kt][nt].q[1] = *((const uint4*)(gp + 8));
            }
        for (int mtile = 0; mtile < 3; ++mtile) {
            const int rbase = mtile * 16;
            FragBF a0, a1;
            const __hip_bfloat16* ap = pbf + (size_t)(c * RP + rbase + lrow) * PBST;
            a0.q[0] = *((const uint4*)(ap +      half * 16));
            a0.q[1] = *((const uint4*)(ap + 8 + half * 16));
            a1.q[0] = *((const uint4*)(ap + 32 +      half * 16));
            a1.q[1] = *((const uint4*)(ap + 32 + 8 + half * 16));
            float part[8] = {0.f, 0.f, 0.f, 0.f, 0.f, 0.f, 0.f, 0.f};
#pragma unroll
            for (int nt = 0; nt < 4; ++nt) {
                v8f acc = {};
                acc = wmma_bf16(a0, g[0][nt], acc);
                acc = wmma_bf16(a1, g[1][nt], acc);
                const int wp = nt * 16 + lrow;
#pragma unroll
                for (int v = 0; v < 8; ++v) {
                    const int r = rbase + v + 8 * half;
                    const float pv = __bfloat162float(pbf[(size_t)(c * RP + r) * PBST + wp]);
                    part[v] += acc[v] * pv;
                }
            }
#pragma unroll
            for (int v = 0; v < 8; ++v) {                            // 16-lane tree sum
                float s = part[v];
                s += __shfl_xor(s, 1, 32);
                s += __shfl_xor(s, 2, 32);
                s += __shfl_xor(s, 4, 32);
                s += __shfl_xor(s, 8, 32);
                if (lrow == 0) s2buf[c * RP + rbase + v + 8 * half] = s;
            }
        }
    }
    __syncthreads();

    // -- phase 4: cosine + masked mean over regions --------------------------
    if (t < NC) {
        const int c = t;
        float acc = 0.f;
        for (int r = 0; r < len; ++r) {
            const float u   = sqrtf(fmaxf(s2buf[c * RP + r], 0.f));  // ||wctx||
            const float w12 = s1buf[c * RP + r] / (u + EPS_F);
            const float w2  = u / (u + EPS_F);
            const float w1  = w1g[i * R_REG + r];
            acc += w12 / fmaxf(w1 * w2, EPS_F);
        }
        out[(size_t)i * N_IMG + capbase + c] = acc / (float)len;
    }
}

// ------------------------------------------------------------- launcher -----
extern "C" void kernel_launch(void* const* d_in, const int* in_sizes, int n_in,
                              void* d_out, int out_size, void* d_ws, size_t ws_size,
                              hipStream_t stream)
{
    (void)in_sizes; (void)n_in; (void)out_size; (void)ws_size;
    const float* imgs     = (const float*)d_in[0];
    const float* caps     = (const float*)d_in[1];
    const int*   img_lens = (const int*)d_in[2];

    char* ws = (char*)d_ws;
    size_t off = 0;
    auto take = [&](size_t bytes) -> void* {
        void* p = ws + off;
        off = (off + bytes + 255) & ~(size_t)255;
        return p;
    };
    const int capsN = N_IMG * W_WORD * D_DIM;     // 26,214,400
    const int imgsN = N_IMG * R_REG * D_DIM;      // 18,874,368
    __hip_bfloat16* caps_bf = (__hip_bfloat16*)take((size_t)capsN * 2);
    __hip_bfloat16* imgs_bf = (__hip_bfloat16*)take((size_t)imgsN * 2);
    __hip_bfloat16* G_bf    = (__hip_bfloat16*)take((size_t)N_IMG * GDIM * GDIM * 2);
    float*          w1g     = (float*)take((size_t)N_IMG * R_REG * 4);

    scan_convert<<<2048, 256, 0, stream>>>(caps, imgs, caps_bf, imgs_bf, capsN, imgsN);
    scan_w1<<<(N_IMG * R_REG) / 8, 256, 0, stream>>>(imgs, w1g, N_IMG * R_REG);
    scan_gram<<<N_IMG, 256, 0, stream>>>(caps_bf, G_bf);

    dim3 g2(64, N_IMG, 1);                               // 64 cap-tiles x 512 images
    scan_i2t_main<<<g2, 256, LDS_TOTAL, stream>>>(caps_bf, imgs_bf, G_bf, w1g,
                                                  img_lens, (float*)d_out);
}
